// Selector_18184891531679
// MI455X (gfx1250) — matrix operand order repaired and checked by reading, not compiled, not run on hardware
//
#include <hip/hip_runtime.h>
#include <hip/hip_bf16.h>
#include <math.h>

typedef __attribute__((ext_vector_type(16))) _Float16 v16h;
typedef __attribute__((ext_vector_type(8)))  float    v8f;

union HF16 { v16h v; uint4 q[2]; };
union V8F  { v8f  v; float f[8]; };

#define B_    32
#define L_    512
#define T_    64
#define Kent_ 8
#define E_    300
#define TE_   16
#define H_    512
#define A_    128
#define D_    512
#define NTAG_ 5
#define XF    352   // E + 2*TE = 332 padded to mult of 32
#define AVF   320   // E = 300 padded to mult of 32
#define G4    2048  // 4*H

__device__ __forceinline__ float sigf(float x) { return 1.0f / (1.0f + __expf(-x)); }

// ---- WMMA fragment loaders (NT layout, row-major over K) -------------------
// A 16x32 f16: lane half h=lane/16; elems 0..15 <-> k = h*16 + e
__device__ __forceinline__ v16h frag_a(const _Float16* base, int ld, int row, int kbase, int lane) {
  int h = lane >> 4;
  const _Float16* p = base + (size_t)row * ld + kbase + h * 16;
  HF16 f;
  f.q[0] = *(const uint4*)(p);
  f.q[1] = *(const uint4*)(p + 8);
  return f.v;
}
// B 32x16 f16 (from W[N,K] row-major): lanes 0-15 hold k=0..15, lanes 16-31 k=16..31; 16 contiguous f16
__device__ __forceinline__ v16h frag_b(const _Float16* base, int ld, int row, int kbase, int lane) {
  int h = lane >> 4;
  const _Float16* p = base + (size_t)row * ld + kbase + h * 16;
  HF16 f;
  f.q[0] = *(const uint4*)(p);
  f.q[1] = *(const uint4*)(p + 8);
  return f.v;
}

// ---- generic NT GEMM: C[M,N] = A[M,K](f16) * W[N,K]^T(f16) + bias, opt ReLU
// block = 256 thr = 8 waves (2x4), block tile 64x128, wave tile 32x32
__global__ __launch_bounds__(256) void k_gemm_nt(float* __restrict__ C,
                                                 const _Float16* __restrict__ A,
                                                 const _Float16* __restrict__ W,
                                                 const float* __restrict__ bias,
                                                 int M, int N, int K, int relu) {
  int lane = threadIdx.x & 31;
  int wv = threadIdx.x >> 5;
  int wm = wv >> 2, wn = wv & 3;
  int m0 = blockIdx.y * 64 + wm * 32;
  int n0 = blockIdx.x * 128 + wn * 32;

  V8F acc[2][2];
#pragma unroll
  for (int a = 0; a < 2; ++a)
#pragma unroll
    for (int b = 0; b < 2; ++b)
#pragma unroll
      for (int v = 0; v < 8; ++v) acc[a][b].f[v] = 0.0f;

  for (int kt = 0; kt < K; kt += 32) {
    v16h fa[2], fb[2];
#pragma unroll
    for (int mt = 0; mt < 2; ++mt) fa[mt] = frag_a(A, K, m0 + mt * 16 + (lane & 15), kt, lane);
#pragma unroll
    for (int nt = 0; nt < 2; ++nt) fb[nt] = frag_b(W, K, n0 + nt * 16 + (lane & 15), kt, lane);
#pragma unroll
    for (int mt = 0; mt < 2; ++mt)
#pragma unroll
      for (int nt = 0; nt < 2; ++nt)
        acc[mt][nt].v = __builtin_amdgcn_wmma_f32_16x16x32_f16(
            false, fa[mt], false, fb[nt], (short)0, acc[mt][nt].v, false, false);
  }

#pragma unroll
  for (int mt = 0; mt < 2; ++mt)
#pragma unroll
    for (int nt = 0; nt < 2; ++nt)
#pragma unroll
      for (int v = 0; v < 8; ++v) {
        int m = m0 + mt * 16 + (lane >> 4) * 8 + v;
        int n = n0 + nt * 16 + (lane & 15);
        float val = acc[mt][nt].f[v] + bias[n];
        if (relu) val = fmaxf(val, 0.0f);
        C[(size_t)m * N + n] = val;
      }
}

// ---- persistent bi-LSTM recurrence -----------------------------------------
// grid.x = 2 (dir 0=fw, 1=bw/reversed), 1024 threads = 32 waves.
// xg layout [S*B, 2048] rows = t*B + b (bias prefolded). h kept f16 in LDS, c in regs.
// Wave w owns cells j = w*16 .. w*16+15 for all 4 gates (tiles at n = q*512 + j).
__global__ __launch_bounds__(1024) void k_lstm(const float* __restrict__ xg0,
                                               const float* __restrict__ xg1,
                                               const _Float16* __restrict__ Wh0,
                                               const _Float16* __restrict__ Wh1,
                                               float* __restrict__ out,
                                               int S, int out_w) {
  int dir = blockIdx.x;
  const float* xg = dir ? xg1 : xg0;
  const _Float16* Wh = dir ? Wh1 : Wh0;
  int col0 = dir ? H_ : 0;
  int rev = dir;

  __shared__ _Float16 hsh[32][520];  // 520-half row stride -> no LDS bank pathologies
  int lane = threadIdx.x & 31;
  int wv = threadIdx.x >> 5;
  int jcol = wv * 16 + (lane & 15);

  for (int i = threadIdx.x; i < 32 * 520; i += 1024) (&hsh[0][0])[i] = (_Float16)0.0f;
  float cst[2][8];
#pragma unroll
  for (int mt = 0; mt < 2; ++mt)
#pragma unroll
    for (int v = 0; v < 8; ++v) cst[mt][v] = 0.0f;
  __syncthreads();

  for (int s = 0; s < S; ++s) {
    int pos = rev ? (S - 1 - s) : s;

    V8F acc[4][2];
#pragma unroll
    for (int q = 0; q < 4; ++q)
#pragma unroll
      for (int mt = 0; mt < 2; ++mt)
#pragma unroll
        for (int v = 0; v < 8; ++v) {
          int bm = mt * 16 + (lane >> 4) * 8 + v;
          acc[q][mt].f[v] = xg[((size_t)pos * B_ + bm) * G4 + q * H_ + jcol];
        }

    for (int kt = 0; kt < H_; kt += 32) {
      v16h fa[2];
#pragma unroll
      for (int mt = 0; mt < 2; ++mt) fa[mt] = frag_a(&hsh[0][0], 520, mt * 16 + (lane & 15), kt, lane);
#pragma unroll
      for (int q = 0; q < 4; ++q) {
        v16h fb = frag_b(Wh, H_, q * H_ + jcol, kt, lane);
        acc[q][0].v = __builtin_amdgcn_wmma_f32_16x16x32_f16(false, fa[0], false, fb, (short)0, acc[q][0].v, false, false);
        acc[q][1].v = __builtin_amdgcn_wmma_f32_16x16x32_f16(false, fa[1], false, fb, (short)0, acc[q][1].v, false, false);
      }
    }

    float hval[2][8];
#pragma unroll
    for (int mt = 0; mt < 2; ++mt)
#pragma unroll
      for (int v = 0; v < 8; ++v) {
        float gi = acc[0][mt].f[v], gf = acc[1][mt].f[v];
        float gg = acc[2][mt].f[v], go = acc[3][mt].f[v];
        float c = sigf(gf) * cst[mt][v] + sigf(gi) * tanhf(gg);
        cst[mt][v] = c;
        hval[mt][v] = sigf(go) * tanhf(c);
      }

    __syncthreads();  // all waves done reading previous h
#pragma unroll
    for (int mt = 0; mt < 2; ++mt)
#pragma unroll
      for (int v = 0; v < 8; ++v) {
        int bm = mt * 16 + (lane >> 4) * 8 + v;
        hsh[bm][jcol] = (_Float16)hval[mt][v];
        out[(size_t)bm * S * out_w + (size_t)pos * out_w + col0 + jcol] = hval[mt][v];
      }
    __syncthreads();  // new h visible
  }
}

// ---- embedding concat -> x_f16 [L*B rows = l*B+b][XF] ----------------------
__global__ void k_build_x(const int* __restrict__ src, const int* __restrict__ ans,
                          const int* __restrict__ endt, const float* __restrict__ semb,
                          const float* __restrict__ aemb, const float* __restrict__ eemb,
                          _Float16* __restrict__ x) {
  long idx = (long)blockIdx.x * blockDim.x + threadIdx.x;
  if (idx >= (long)L_ * B_ * XF) return;
  int r = (int)(idx / XF), c = (int)(idx % XF);
  int b = r & (B_ - 1), l = r >> 5;
  float v = 0.0f;
  if (c < E_)            v = semb[(size_t)src[b * L_ + l] * E_ + c];
  else if (c < E_ + TE_) v = aemb[ans[b * L_ + l] * TE_ + (c - E_)];
  else if (c < E_ + 2 * TE_) v = eemb[endt[b * L_ + l] * TE_ + (c - E_ - TE_)];
  x[idx] = (_Float16)v;
}

// ---- masked entity mean -> avg_f16 [T*B rows = t*B+b][AVF] -----------------
__global__ void k_avg(const int* __restrict__ ent, const float* __restrict__ semb,
                      _Float16* __restrict__ avg) {
  long idx = (long)blockIdx.x * blockDim.x + threadIdx.x;
  if (idx >= (long)T_ * B_ * AVF) return;
  int r = (int)(idx / AVF), c = (int)(idx % AVF);
  int b = r & (B_ - 1), t = r >> 5;
  float val = 0.0f;
  if (c < E_) {
    float s = 0.0f, cnt = 0.0f;
#pragma unroll
    for (int k = 0; k < Kent_; ++k) {
      int tok = ent[((size_t)b * T_ + t) * Kent_ + k];
      if (tok > 0) { s += semb[(size_t)tok * E_ + c]; cnt += 1.0f; }
    }
    val = s / fmaxf(cnt, 1.0f);
  }
  avg[idx] = (_Float16)val;
}

// ---- f32 -> f16 convert with pad -------------------------------------------
__global__ void k_cvt(_Float16* __restrict__ dst, const float* __restrict__ src,
                      int rows, int cols, int sld, int dld) {
  long idx = (long)blockIdx.x * blockDim.x + threadIdx.x;
  if (idx >= (long)rows * dld) return;
  int r = (int)(idx / dld), c = (int)(idx % dld);
  dst[idx] = (c < cols) ? (_Float16)src[(size_t)r * sld + c] : (_Float16)0.0f;
}

// ---- fused Bahdanau scores + softmax + ctx; grid = (T/16, B), 1024 thr -----
__global__ __launch_bounds__(1024) void k_attn(const float* __restrict__ q,
                                               const float* __restrict__ kp,
                                               const float* __restrict__ vvec,
                                               const float* __restrict__ enc,
                                               float* __restrict__ feat) {
  int b = blockIdx.y, t0 = blockIdx.x * 16;
  __shared__ float sc[16][L_];    // 32 KB
  __shared__ float qs[16][A_];    // 8 KB
  __shared__ float kc[32][A_];    // 16 KB
  __shared__ float vs[A_];
  int tid = threadIdx.x;

  for (int i = tid; i < 16 * A_; i += 1024)
    qs[i / A_][i % A_] = q[((size_t)(b * T_ + t0 + i / A_)) * A_ + (i % A_)];
  if (tid < A_) vs[tid] = vvec[tid];
  __syncthreads();

  for (int lc = 0; lc < L_; lc += 32) {
    for (int i = tid; i < 32 * A_; i += 1024)
      kc[i / A_][i % A_] = kp[((size_t)(b * L_ + lc + i / A_)) * A_ + (i % A_)];
    __syncthreads();
    if (tid < 512) {
      int tt = tid >> 5, lr = tid & 31;
      float s = 0.0f;
#pragma unroll 8
      for (int a = 0; a < A_; ++a) s += tanhf(qs[tt][a] + kc[lr][a]) * vs[a];
      sc[tt][lc + lr] = s;
    }
    __syncthreads();
  }

  if (tid < 16) {  // row softmax
    float mx = -1e30f;
    for (int l = 0; l < L_; ++l) mx = fmaxf(mx, sc[tid][l]);
    float sum = 0.0f;
    for (int l = 0; l < L_; ++l) { float e = __expf(sc[tid][l] - mx); sc[tid][l] = e; sum += e; }
    float inv = 1.0f / sum;
    for (int l = 0; l < L_; ++l) sc[tid][l] *= inv;
  }
  __syncthreads();

  int d = tid;  // 1024 threads <-> 1024 enc dims
  float acc[16];
#pragma unroll
  for (int tt = 0; tt < 16; ++tt) acc[tt] = 0.0f;
  for (int l = 0; l < L_; ++l) {
    float e = enc[((size_t)b * L_ + l) * (2 * H_) + d];
#pragma unroll
    for (int tt = 0; tt < 16; ++tt) acc[tt] += sc[tt][l] * e;
  }
#pragma unroll
  for (int tt = 0; tt < 16; ++tt)
    feat[((size_t)(b * T_ + t0 + tt)) * 2048 + 1024 + d] = acc[tt];
}

// ---- tiny output head: logits[r][n] = dense[r,:]·outW[n,:] + b -------------
__global__ void k_out(const float* __restrict__ dense, const float* __restrict__ W,
                      const float* __restrict__ bias, float* __restrict__ out) {
  int idx = blockIdx.x * blockDim.x + threadIdx.x;
  if (idx >= B_ * T_ * NTAG_) return;
  int r = idx / NTAG_, n = idx % NTAG_;
  float s = bias[n];
  for (int k = 0; k < D_; ++k) s += dense[(size_t)r * D_ + k] * W[(size_t)n * D_ + k];
  out[idx] = s;
}

extern "C" void kernel_launch(void* const* d_in, const int* in_sizes, int n_in,
                              void* d_out, int out_size, void* d_ws, size_t ws_size,
                              hipStream_t stream) {
  (void)in_sizes; (void)n_in; (void)out_size; (void)ws_size;
  const int*   src      = (const int*)d_in[0];
  const int*   anst     = (const int*)d_in[2];
  const int*   endt     = (const int*)d_in[3];
  const int*   ent      = (const int*)d_in[4];
  const float* semb     = (const float*)d_in[9];
  const float* aemb     = (const float*)d_in[10];
  const float* eemb     = (const float*)d_in[11];
  const float* eWih0    = (const float*)d_in[12];
  const float* eWhh0    = (const float*)d_in[13];
  const float* eb0      = (const float*)d_in[14];
  const float* eWih1    = (const float*)d_in[15];
  const float* eWhh1    = (const float*)d_in[16];
  const float* eb1      = (const float*)d_in[17];
  const float* cWih0    = (const float*)d_in[18];
  const float* cWhh0    = (const float*)d_in[19];
  const float* cb0      = (const float*)d_in[20];
  const float* cWih1    = (const float*)d_in[21];
  const float* cWhh1    = (const float*)d_in[22];
  const float* cb1      = (const float*)d_in[23];
  const float* Wq       = (const float*)d_in[24];
  const float* bq       = (const float*)d_in[25];
  const float* Wk       = (const float*)d_in[26];
  const float* bk       = (const float*)d_in[27];
  const float* av       = (const float*)d_in[28];
  const float* dW       = (const float*)d_in[29];
  const float* db       = (const float*)d_in[30];
  const float* oW       = (const float*)d_in[31];
  const float* ob       = (const float*)d_in[32];

  size_t off = 0;
  auto carve = [&](size_t bytes) -> char* {
    char* p = (char*)d_ws + off;
    off += (bytes + 255) & ~(size_t)255;
    return p;
  };
  _Float16* x16    = (_Float16*)carve((size_t)L_ * B_ * XF * 2);
  _Float16* Wih0h  = (_Float16*)carve((size_t)G4 * XF * 2);
  _Float16* Wih1h  = (_Float16*)carve((size_t)G4 * XF * 2);
  _Float16* Whh0h  = (_Float16*)carve((size_t)G4 * H_ * 2);
  _Float16* Whh1h  = (_Float16*)carve((size_t)G4 * H_ * 2);
  float*    xgf    = (float*)carve((size_t)L_ * B_ * G4 * 4);
  float*    xgb    = (float*)carve((size_t)L_ * B_ * G4 * 4);
  float*    enc    = (float*)carve((size_t)B_ * L_ * 2 * H_ * 4);
  _Float16* avg16  = (_Float16*)carve((size_t)T_ * B_ * AVF * 2);
  _Float16* cWih0h = (_Float16*)carve((size_t)G4 * AVF * 2);
  _Float16* cWih1h = (_Float16*)carve((size_t)G4 * AVF * 2);
  _Float16* cWhh0h = (_Float16*)carve((size_t)G4 * H_ * 2);
  _Float16* cWhh1h = (_Float16*)carve((size_t)G4 * H_ * 2);
  float*    xgc0   = (float*)carve((size_t)T_ * B_ * G4 * 4);
  float*    xgc1   = (float*)carve((size_t)T_ * B_ * G4 * 4);
  float*    feat   = (float*)carve((size_t)B_ * T_ * 2048 * 4);
  _Float16* enc16  = (_Float16*)carve((size_t)B_ * L_ * 1024 * 2);
  _Float16* out16  = (_Float16*)carve((size_t)B_ * T_ * 1024 * 2);
  _Float16* Wq16   = (_Float16*)carve((size_t)A_ * 1024 * 2);
  _Float16* Wk16   = (_Float16*)carve((size_t)A_ * 1024 * 2);
  float*    qb     = (float*)carve((size_t)B_ * T_ * A_ * 4);
  float*    kpb    = (float*)carve((size_t)B_ * L_ * A_ * 4);
  _Float16* feat16 = (_Float16*)carve((size_t)B_ * T_ * 2048 * 2);
  _Float16* dW16   = (_Float16*)carve((size_t)D_ * 2048 * 2);
  float*    dense  = (float*)carve((size_t)B_ * T_ * D_ * 4);

  auto cvt = [&](_Float16* dst, const float* s, int rows, int cols, int sld, int dld) {
    long total = (long)rows * dld;
    k_cvt<<<(unsigned)((total + 255) / 256), 256, 0, stream>>>(dst, s, rows, cols, sld, dld);
  };
  auto gemm = [&](float* C, const _Float16* A, const _Float16* W, const float* bias,
                  int M, int N, int K, int relu) {
    dim3 g(N / 128, M / 64);
    k_gemm_nt<<<g, 256, 0, stream>>>(C, A, W, bias, M, N, K, relu);
  };

  // stage 0: embeddings + weight conversion
  k_build_x<<<(unsigned)(((long)L_ * B_ * XF + 255) / 256), 256, 0, stream>>>(src, anst, endt, semb, aemb, eemb, x16);
  k_avg<<<(unsigned)(((long)T_ * B_ * AVF + 255) / 256), 256, 0, stream>>>(ent, semb, avg16);
  cvt(Wih0h, eWih0, G4, 332, 332, XF);
  cvt(Wih1h, eWih1, G4, 332, 332, XF);
  cvt(Whh0h, eWhh0, G4, H_, H_, H_);
  cvt(Whh1h, eWhh1, G4, H_, H_, H_);
  cvt(cWih0h, cWih0, G4, E_, E_, AVF);
  cvt(cWih1h, cWih1, G4, E_, E_, AVF);
  cvt(cWhh0h, cWhh0, G4, H_, H_, H_);
  cvt(cWhh1h, cWhh1, G4, H_, H_, H_);
  cvt(Wq16, Wq, A_, 1024, 1024, 1024);
  cvt(Wk16, Wk, A_, 1024, 1024, 1024);
  cvt(dW16, dW, D_, 2048, 2048, 2048);

  // stage 1: input projections (WMMA GEMMs, bias folded)
  gemm(xgf, x16, Wih0h, eb0, L_ * B_, G4, XF, 0);
  gemm(xgb, x16, Wih1h, eb1, L_ * B_, G4, XF, 0);
  gemm(xgc0, avg16, cWih0h, cb0, T_ * B_, G4, AVF, 0);
  gemm(xgc1, avg16, cWih1h, cb1, T_ * B_, G4, AVF, 0);

  // stage 2: recurrences (both directions concurrently per launch)
  k_lstm<<<2, 1024, 0, stream>>>(xgf, xgb, Whh0h, Whh1h, enc, L_, 2 * H_);
  k_lstm<<<2, 1024, 0, stream>>>(xgc0, xgc1, cWhh0h, cWhh1h, feat, T_, 2048);

  // stage 3: attention projections
  cvt(enc16, enc, B_ * L_, 1024, 1024, 1024);
  cvt(out16, feat, B_ * T_, 1024, 2048, 1024);
  gemm(kpb, enc16, Wk16, bk, B_ * L_, A_, 1024, 0);
  gemm(qb, out16, Wq16, bq, B_ * T_, A_, 1024, 0);

  // stage 4: fused scores/softmax/ctx -> feat[:, 1024:2048]
  k_attn<<<dim3(T_ / 16, B_), 1024, 0, stream>>>(qb, kpb, av, enc, feat);

  // stage 5: dense + output head
  cvt(feat16, feat, B_ * T_, 2048, 2048, 2048);
  gemm(dense, feat16, dW16, db, B_ * T_, D_, 2048, 1);
  k_out<<<(B_ * T_ * NTAG_ + 255) / 256, 256, 0, stream>>>(dense, oW, ob, (float*)d_out);
}
